// ClusteredGraphReconstructor_27917287424628
// MI455X (gfx1250) — matrix operand order repaired and checked by reading, not compiled, not run on hardware
//
#include <hip/hip_runtime.h>
#include <hip/hip_bf16.h>
#include <math.h>

// CDNA5 / gfx1250, wave32.
typedef __attribute__((ext_vector_type(16))) _Float16     v16h;
typedef __attribute__((ext_vector_type(8)))  float        v8f;
typedef __attribute__((ext_vector_type(4)))  unsigned int v4u;
typedef __attribute__((ext_vector_type(8)))  int          v8i;
typedef __attribute__((ext_vector_type(4)))  int          v4i;

#define N_CLUSTERS  32
#define N_RELATIONS 8
// Pre-swizzled weight buffer: 8 relations x 2 row-tiles x 32 lanes x 16 halves
#define WLAY_ELEMS  (N_RELATIONS * 2 * 32 * 16)   // 8192 halves = 16 KB

__device__ __forceinline__ float sigmoidf_fast(float x) {
  return 1.0f / (1.0f + __expf(-x));
}

// Hard-concrete gate, eval mode: pre = clip(la/(2/3), -2, 2);
// z = sigmoid(pre)*1.2 - 0.1; clip(z, 0, 1).
__device__ __forceinline__ float hc_gate(float la) {
  float pre = fminf(fmaxf(la * 1.5f, -2.0f), 2.0f);
  float z = sigmoidf_fast(pre) * 1.2f - 0.1f;
  return fminf(fmaxf(z, 0.0f), 1.0f);
}

// Precompute W[r] = sigmoid(logits)*gate(log_alpha) in f16, permuted into the
// exact V_WMMA_F32_16X16X32_F16 A-operand register layout (ISA 7.12.2,
// "16-bit A-Matrix 16x32"):
//   lane L (group g=L/16): row M = L%16 of tile t (rows i = t*16+M)
//   element e=2v+h of the lane's v16h: K = (v<4?0:16) + (v&3)*2 + h + g*8
__global__ void prep_weights(const float* __restrict__ logits,
                             const float* __restrict__ logalpha,
                             _Float16* __restrict__ wlay) {
  int id = blockIdx.x * blockDim.x + threadIdx.x;
  if (id >= WLAY_ELEMS) return;
  int e = id & 15;          // element within lane's v16h
  int L = (id >> 4) & 31;   // lane
  int t = (id >> 9) & 1;    // row tile (i 0..15 vs 16..31)
  int r = id >> 10;         // relation
  int g = L >> 4;
  int v = e >> 1;
  int h = e & 1;
  int j = ((v < 4) ? 0 : 16) + (v & 3) * 2 + h + g * 8;  // K index (dst dim)
  int i = t * 16 + (L & 15);                              // output row
  int off = (r * N_CLUSTERS + i) * N_CLUSTERS + j;
  float w = sigmoidf_fast(logits[off]) * hc_gate(logalpha[off]);
  wlay[id] = (_Float16)w;
}

// One wave processes 16 edges per batch:
//   B[32x16]   : column = edge, rows = dst vector (f16)
//   Y = W_r * B via 2 x v_wmma_f32_16x16x32_f16 per relation (2 row tiles)
//   out[e] = sum_i src[e][i] * Y[i][e]  (+ bias), relation-selected per edge.
// Weights are DMA'd into LDS once per block by the Tensor Data Mover
// (TENSOR_LOAD_TO_LDS, tracked by TENSORcnt); an opaque asm memory barrier
// tells the optimizer that LDS was written (the intrinsic carries no pointer
// alias info, which round 3 showed lets LLVM fold the LDS loads to undef).
__global__ __launch_bounds__(256) void edge_logits_wmma(
    const float* __restrict__ assign,     // [N, 32] f32
    const int*   __restrict__ eidx,       // [2, E]
    const int*   __restrict__ etype,      // [E]
    const _Float16* __restrict__ wlay,    // pre-swizzled weights (global)
    const float* __restrict__ bias,       // [8]
    float* __restrict__ out,              // [E]
    int E) {
  __shared__ _Float16 sA[WLAY_ELEMS];     // 16 KB of 320 KB/WGP, LDS offset 0

  // --- TDM stage: one wave issues a tensor DMA of the 16 KB weight image ---
  // D# (CDNA5 ISA): 2D tensor, 4096 x 1 elements of 4 bytes,
  // tile = whole tensor, lds_addr = 0 (sA is the only LDS object).
  if (threadIdx.x < 32) {
    unsigned long long ga = (unsigned long long)(__UINTPTR_TYPE__)wlay;
    v4u g0;
    g0[0] = 1u;                                   // count=1, user descriptor
    g0[1] = 0u;                                   // lds_addr = 0
    g0[2] = (unsigned int)(ga & 0xFFFFFFFFu);     // global_addr[31:0]
    g0[3] = (unsigned int)((ga >> 32) & 0x01FFFFFFu) | 0x80000000u; // [56:32] | type=2
    v8i g1;
    g1[0] = (2 << 16);        // workgroup_mask=0 (not in cluster), data_size=4B
    g1[1] = (4096 << 16);     // tensor_dim0 = 4096 (low 16 in [31:16])
    g1[2] = (1 << 16);        // tensor_dim0 hi = 0, tensor_dim1 = 1
    g1[3] = (4096 << 16);     // tensor_dim1 hi = 0, tile_dim0 = 4096
    g1[4] = 1;                // tile_dim1 = 1, tile_dim2 = 0 (unused)
    g1[5] = 4096;             // tensor_dim0_stride
    g1[6] = 0;
    g1[7] = 0;
    v4i gz = {0, 0, 0, 0};    // groups 2/3: dims 2..4 unused (zeros)
#if defined(__clang_major__) && (__clang_major__ >= 23)
    v8i gz8 = {0, 0, 0, 0, 0, 0, 0, 0};
    __builtin_amdgcn_tensor_load_to_lds(g0, g1, gz, gz, gz8, 0);
#else
    __builtin_amdgcn_tensor_load_to_lds(g0, g1, gz, gz, 0);
#endif
    __builtin_amdgcn_s_wait_tensorcnt(0);         // s_wait_tensorcnt 0
  }
  __syncthreads();
  // Opaque writer: prevents LLVM from treating sA as never-written (the TDM
  // intrinsic has no pointer operand the alias analysis can see).
  {
    _Float16* p = sA;
    asm volatile("" : "+r"(p) : : "memory");
  }

  const int lane = threadIdx.x & 31;
  const int g    = lane >> 4;   // half-wave group
  const int el   = lane & 15;   // edge slot within batch
  const int wid    = (blockIdx.x * blockDim.x + threadIdx.x) >> 5;
  const int nwaves = (gridDim.x * blockDim.x) >> 5;

  const _Float16* aBase = sA + lane * 16;   // + (2r+t)*32*16 per matrix

  const int nb = (E + 15) >> 4;
  for (int b = wid; b < nb; b += nwaves) {
    int idx  = b * 16 + el;
    int edge = idx < E ? idx : (E - 1);  // clamp: keep EXEC all-ones for WMMA
    int srcn = eidx[edge];               // edge_index[0][edge]
    int dstn = eidx[E + edge];           // edge_index[1][edge]
    int r_e  = etype[edge];

    // Prefetch next batch's index lines (global_prefetch_b8).
    int bn = b + nwaves;
    if (bn < nb) {
      int pidx = bn * 16 + el;
      __builtin_prefetch(&eidx[pidx], 0, 0);
      __builtin_prefetch(&eidx[E + pidx], 0, 0);
      __builtin_prefetch(&etype[pidx], 0, 0);
    }

    // B operand: this lane supplies K = g*8 .. g*8+7 and 16+g*8 .. 16+g*8+7 of column `el`.
    const float* drow = assign + dstn * N_CLUSTERS + g * 8;
    v16h Bm;
#pragma unroll
    for (int k = 0; k < 16; k += 4) {
      float4 q = *(const float4*)(drow + (k < 8 ? k : k + 8));
      Bm[k + 0] = (_Float16)q.x;
      Bm[k + 1] = (_Float16)q.y;
      Bm[k + 2] = (_Float16)q.z;
      Bm[k + 3] = (_Float16)q.w;
    }

    // src in the C/D layout: VGPR v of tile t corresponds to i = t*16 + v + 8*g.
    const float* srow = assign + srcn * N_CLUSTERS + g * 8;
    float4 s00 = *(const float4*)(srow +  0);
    float4 s01 = *(const float4*)(srow +  4);
    float4 s10 = *(const float4*)(srow + 16);
    float4 s11 = *(const float4*)(srow + 20);
    float S0[8] = {s00.x, s00.y, s00.z, s00.w, s01.x, s01.y, s01.z, s01.w};
    float S1[8] = {s10.x, s10.y, s10.z, s10.w, s11.x, s11.y, s11.z, s11.w};

    float val = 0.0f;
#pragma unroll 1   // keep ONE accumulator pair live -> low VGPR pressure
    for (int r = 0; r < N_RELATIONS; ++r) {
      v16h Am0 = *(const v16h*)(aBase + (2 * r + 0) * (32 * 16));
      v16h Am1 = *(const v16h*)(aBase + (2 * r + 1) * (32 * 16));
      v8f z = {};
      v8f acc0 = __builtin_amdgcn_wmma_f32_16x16x32_f16(
          false, Am0, false, Bm, (short)0, z, false, false);
      v8f acc1 = __builtin_amdgcn_wmma_f32_16x16x32_f16(
          false, Am1, false, Bm, (short)0, z, false, false);
      float pr = 0.0f;
#pragma unroll
      for (int v = 0; v < 8; ++v) pr += acc0[v] * S0[v] + acc1[v] * S1[v];
      val += (r == r_e) ? pr : 0.0f;  // per-edge relation select
    }

    // Lane L and L+16 hold the two half-sums (rows v and v+8) of edge `el`.
    val += __shfl_xor(val, 16, 32);
    if (g == 0 && idx < E) out[idx] = val + bias[r_e];
  }
}

extern "C" void kernel_launch(void* const* d_in, const int* in_sizes, int n_in,
                              void* d_out, int out_size, void* d_ws, size_t ws_size,
                              hipStream_t stream) {
  const float* assign   = (const float*)d_in[0];     // assignments [N,32]
  const int*   eidx     = (const int*)d_in[1];       // edge_index [2,E]
  const int*   etype    = (const int*)d_in[2];       // edge_type [E]
  const float* logits   = (const float*)d_in[3];     // [8,32,32]
  const float* logalpha = (const float*)d_in[4];     // [8,32,32]
  const float* bias     = (const float*)d_in[5];     // [8]
  float*       out      = (float*)d_out;             // [E]
  _Float16*    wlay     = (_Float16*)d_ws;           // 16 KB scratch

  const int E = in_sizes[2];  // number of edges

  // Tiny prep: 8192 gated weights -> pre-swizzled f16 A-operand layout.
  prep_weights<<<(WLAY_ELEMS + 255) / 256, 256, 0, stream>>>(logits, logalpha, wlay);

  // Main pass: grid-stride over 16-edge batches (E/16 = 31250 batches).
  edge_logits_wmma<<<1024, 256, 0, stream>>>(assign, eidx, etype, wlay, bias, out, E);
}
